// CommonRNN_24927990186128
// MI455X (gfx1250) — matrix operand order repaired and checked by reading, not compiled, not run on hardware
//
#include <hip/hip_runtime.h>
#include <hip/hip_bf16.h>

// Problem constants (from reference): B=64, S=512, IN=512, H=512, L=2
#define Bb  64
#define Ss  512
#define INs 512
#define Hh  512
#define Gg  2048   // 4*H gate width

typedef __attribute__((ext_vector_type(16))) __bf16 v16bf;
typedef __attribute__((ext_vector_type(8)))  __bf16 v8bf;
typedef __attribute__((ext_vector_type(8)))  float  v8f;

union V16u { v16bf v; v8bf h[2]; };

// f32 -> bf16 round-to-nearest-even
__device__ __forceinline__ __bf16 f2bf(float x) {
  unsigned u = __builtin_bit_cast(unsigned, x);
  u += 0x7FFFu + ((u >> 16) & 1u);
  unsigned short s = (unsigned short)(u >> 16);
  return __builtin_bit_cast(__bf16, s);
}

__device__ __forceinline__ float sigm(float x) { return 1.0f / (1.0f + __expf(-x)); }

// A-matrix fragment (16x32 bf16, row-major source, row = one lane):
// lanes 0-15 hold row m: K = [k0+0..15]; lanes 16-31: K = [k0+16..31]
__device__ __forceinline__ v16bf load_a16(const __bf16* __restrict__ row, int k0, int half) {
  const __bf16* p = row + k0 + half * 16;
  V16u u;
  u.h[0] = *(const v8bf*)p;
  u.h[1] = *(const v8bf*)(p + 8);
  return u.v;
}

// ---------------- elementwise conversion kernels ----------------
__global__ void __launch_bounds__(256) cvt_w(const float* __restrict__ s,
                                             __bf16* __restrict__ d, int n) {
  int i = blockIdx.x * 256 + threadIdx.x;
  if (i < n) d[i] = f2bf(s[i]);
}

// inputs [B,S,IN] f32  ->  x0 [S,B,IN] bf16
__global__ void __launch_bounds__(256) cvt_in(const float* __restrict__ s,
                                              __bf16* __restrict__ d) {
  int i = blockIdx.x * 256 + threadIdx.x;       // i = t*B*IN + b*IN + k
  int k = i & (INs - 1);
  int b = (i >> 9) & (Bb - 1);
  int t = i >> 15;
  d[i] = f2bf(s[((size_t)b * Ss + t) * INs + k]);
}

// zero h ping-pong buffers (2 dirs x 2 parities x 64 x 512 bf16 = 65536 ints)
// and the 2 per-direction barrier counters. Runs before every scan launch so
// graph replays are deterministic.
__global__ void __launch_bounds__(256) prep_scan(int* __restrict__ hg_int,
                                                 int* __restrict__ cnt) {
  int i = blockIdx.x * 256 + threadIdx.x;
  if (i < 65536) hg_int[i] = 0;
  if (i < 2) cnt[i] = 0;
}

// ---------------- input projection GEMM: xg = X @ Wih^T ----------------
// X: [S*B, K] bf16 row-major.  W: [2048, K] bf16 row-major (W row n = col n of W^T).
// grid: (M/128, 2048/64, 2 dirs), block 256 (8 waves, each 16 rows x 64 cols,
// A-fragment reused across 4 column tiles).
__global__ void __launch_bounds__(256)
proj_gemm(const __bf16* __restrict__ X, int K,
          const __bf16* __restrict__ Wfw, const __bf16* __restrict__ Wbw,
          float* __restrict__ xgfw, float* __restrict__ xgbw) {
  const int w    = threadIdx.x >> 5;
  const int lane = threadIdx.x & 31;
  const int half = lane >> 4;
  const int ln   = lane & 15;
  const int m0   = blockIdx.x * 128 + w * 16;
  const int n0   = blockIdx.y * 64;
  const __bf16* W  = blockIdx.z ? Wbw  : Wfw;
  float*        xg = blockIdx.z ? xgbw : xgfw;

  const __bf16* arow = X + (size_t)(m0 + ln) * K;

  v8f acc[4];
#pragma unroll
  for (int j = 0; j < 4; ++j)
#pragma unroll
    for (int r = 0; r < 8; ++r) acc[j][r] = 0.f;

  for (int k0 = 0; k0 < K; k0 += 32) {
    v16bf a = load_a16(arow, k0, half);
#pragma unroll
    for (int j = 0; j < 4; ++j) {
      const __bf16* bp = W + (size_t)(n0 + j * 16 + ln) * K + k0 + half * 16;
      v16bf b = *(const v16bf*)bp;
      acc[j] = __builtin_amdgcn_wmma_f32_16x16x32_bf16(false, a, false, b,
                                                       (short)0, acc[j], false, false);
    }
  }
#pragma unroll
  for (int j = 0; j < 4; ++j) {
#pragma unroll
    for (int r = 0; r < 8; ++r) {
      int row = m0 + half * 8 + r;      // C/D layout: lanes 0-15 -> M=r, 16-31 -> M=8+r
      xg[(size_t)row * Gg + n0 + j * 16 + ln] = acc[j][r];
    }
  }
}

// ---------------- recurrent scan, 16 workgroups per direction ----------------
// grid = 32 blocks x 1024 threads. dir = blockIdx.x>>4, slice g = blockIdx.x&15
// owns h columns [g*32, g*32+32). Per step: each wave computes one 16x16 gate
// tile (K=512 -> 16 WMMAs), gates staged in LDS, statically-owned threads do the
// cell update (c stays in VGPRs), new h (bf16) goes to a parity-swapped global
// buffer, then a device-scope atomic barrier (monotonic counter + s_sleep spin)
// synchronizes the 16 slices.
__global__ void __launch_bounds__(1024)
lstm_scan_mb(const float* __restrict__ xgfw, const float* __restrict__ xgbw,
             const __bf16* __restrict__ Whhfw, const __bf16* __restrict__ Whhbw,
             const float* __restrict__ bihfw, const float* __restrict__ bhhfw,
             const float* __restrict__ bihbw, const float* __restrict__ bhhbw,
             const float* __restrict__ mask,       // [B,S]
             __bf16* __restrict__ hglob,           // [2][2][64][512] bf16 ping-pong
             int* __restrict__ cnt,                // [2] barrier counters
             __bf16* __restrict__ out_bf,          // layer0: [S,B,2H] bf16 (else null)
             float*  __restrict__ out_f32,         // layer1: [B,S,2H] f32  (else null)
             float* __restrict__ hn, float* __restrict__ cn) {  // [B,2H] slices
  __shared__ __align__(16) float gbuf[Bb * 128];   // 32KB: gates for this 32-col slice

  const int dir = blockIdx.x >> 4;
  const int g   = blockIdx.x & 15;
  const float*  xg  = dir ? xgbw  : xgfw;
  const __bf16* Whh = dir ? Whhbw : Whhfw;
  const float*  bih = dir ? bihbw : bihfw;
  const float*  bhh = dir ? bhhbw : bhhfw;
  __bf16* hbase = hglob + (size_t)dir * 2 * Bb * Hh;
  int* mycnt = cnt + dir;

  const int tid  = threadIdx.x;
  const int w    = tid >> 5;
  const int lane = tid & 31;
  const int half = lane >> 4;
  const int ln   = lane & 15;
  const int m0   = (w & 3) * 16;            // row tile
  const int gi   = w >> 2;                  // 0..7
  const int q    = gi >> 1;                 // gate (i,f,g,o)
  const int ct   = gi & 1;                  // col tile within slice
  const int n    = q * Hh + g * 32 + ct * 16 + ln;   // global gate column
  const int lc   = q * 32 + ct * 16 + ln;            // LDS gate column

  const float bsum = bih[n] + bhh[n];

  // cell-update ownership: elements e0/e1 of the 64x32 slice, fixed across steps
  const int e0 = tid, e1 = tid + 1024;
  const int row0 = e0 >> 5, cl0 = e0 & 31;
  const int row1 = e1 >> 5, cl1 = e1 & 31;
  float c_a = 0.f, c_b = 0.f, hl_a = 0.f, hl_b = 0.f;

  for (int p = 0; p < Ss; ++p) {
    const int t = dir ? (Ss - 1 - p) : p;
    const __bf16* h_in  = hbase + (size_t)(p & 1) * (Bb * Hh);
    __bf16*       h_out = hbase + (size_t)((p + 1) & 1) * (Bb * Hh);

    // ---- gate GEMM: acc = xg_t + bias + h @ Whh^T  (one tile per wave) ----
    v8f acc;
#pragma unroll
    for (int r = 0; r < 8; ++r) {
      int row = m0 + half * 8 + r;
      acc[r] = xg[((size_t)t * Bb + row) * Gg + n] + bsum;
    }
    const __bf16* arow = h_in + (size_t)(m0 + ln) * Hh;
    const __bf16* brow = Whh + (size_t)n * Hh + half * 16;
    for (int k0 = 0; k0 < Hh; k0 += 32) {
      v16bf a = load_a16(arow, k0, half);
      v16bf b = *(const v16bf*)(brow + k0);
      acc = __builtin_amdgcn_wmma_f32_16x16x32_bf16(false, a, false, b,
                                                    (short)0, acc, false, false);
    }
#pragma unroll
    for (int r = 0; r < 8; ++r) {
      int row = m0 + half * 8 + r;
      gbuf[row * 128 + lc] = acc[r];
    }
    __syncthreads();

    // ---- cell update (gate order i,f,g,o) + mask, 2 elements per thread ----
    {
      float iv = sigm(gbuf[row0 * 128 + cl0]);
      float fv = sigm(gbuf[row0 * 128 + 32 + cl0]);
      float gv = tanhf(gbuf[row0 * 128 + 64 + cl0]);
      float ov = sigm(gbuf[row0 * 128 + 96 + cl0]);
      float cnew = fv * c_a + iv * gv;
      float hnew = ov * tanhf(cnew);
      float mv = mask[(size_t)row0 * Ss + t];
      float h2 = hnew * mv; c_a = cnew * mv; hl_a = h2;
      int col = g * 32 + cl0;
      h_out[row0 * Hh + col] = f2bf(h2);
      if (out_bf) out_bf[((size_t)t * Bb + row0) * (2 * Hh) + dir * Hh + col] = f2bf(h2);
      else        out_f32[((size_t)row0 * Ss + t) * (2 * Hh) + dir * Hh + col] = h2;
    }
    {
      float iv = sigm(gbuf[row1 * 128 + cl1]);
      float fv = sigm(gbuf[row1 * 128 + 32 + cl1]);
      float gv = tanhf(gbuf[row1 * 128 + 64 + cl1]);
      float ov = sigm(gbuf[row1 * 128 + 96 + cl1]);
      float cnew = fv * c_b + iv * gv;
      float hnew = ov * tanhf(cnew);
      float mv = mask[(size_t)row1 * Ss + t];
      float h2 = hnew * mv; c_b = cnew * mv; hl_b = h2;
      int col = g * 32 + cl1;
      h_out[row1 * Hh + col] = f2bf(h2);
      if (out_bf) out_bf[((size_t)t * Bb + row1) * (2 * Hh) + dir * Hh + col] = f2bf(h2);
      else        out_f32[((size_t)row1 * Ss + t) * (2 * Hh) + dir * Hh + col] = h2;
    }

    // ---- device-scope barrier across the 16 slices of this direction ----
    __threadfence();
    __syncthreads();          // also separates gbuf reads from next step's writes
    if (tid == 0) {
      __hip_atomic_fetch_add(mycnt, 1, __ATOMIC_RELEASE, __HIP_MEMORY_SCOPE_AGENT);
      const int target = 16 * (p + 1);
      while (__hip_atomic_load(mycnt, __ATOMIC_ACQUIRE, __HIP_MEMORY_SCOPE_AGENT) < target)
        __builtin_amdgcn_s_sleep(1);
    }
    __syncthreads();
    __threadfence();
  }

  // final (post-mask) states
  {
    int col = g * 32 + cl0;
    hn[(size_t)row0 * (2 * Hh) + dir * Hh + col] = hl_a;
    cn[(size_t)row0 * (2 * Hh) + dir * Hh + col] = c_a;
  }
  {
    int col = g * 32 + cl1;
    hn[(size_t)row1 * (2 * Hh) + dir * Hh + col] = hl_b;
    cn[(size_t)row1 * (2 * Hh) + dir * Hh + col] = c_b;
  }
}

extern "C" void kernel_launch(void* const* d_in, const int* in_sizes, int n_in,
                              void* d_out, int out_size, void* d_ws, size_t ws_size,
                              hipStream_t stream) {
  const float* inputs  = (const float*)d_in[0];
  const float* mask    = (const float*)d_in[1];
  const float* fw0_Wih = (const float*)d_in[2];
  const float* fw0_Whh = (const float*)d_in[3];
  const float* fw0_bih = (const float*)d_in[4];
  const float* fw0_bhh = (const float*)d_in[5];
  const float* bw0_Wih = (const float*)d_in[6];
  const float* bw0_Whh = (const float*)d_in[7];
  const float* bw0_bih = (const float*)d_in[8];
  const float* bw0_bhh = (const float*)d_in[9];
  const float* fw1_Wih = (const float*)d_in[10];
  const float* fw1_Whh = (const float*)d_in[11];
  const float* fw1_bih = (const float*)d_in[12];
  const float* fw1_bhh = (const float*)d_in[13];
  const float* bw1_Wih = (const float*)d_in[14];
  const float* bw1_Whh = (const float*)d_in[15];
  const float* bw1_bih = (const float*)d_in[16];
  const float* bw1_bhh = (const float*)d_in[17];

  // ---- workspace layout (all offsets 256B aligned) ----
  char* ws = (char*)d_ws;
  __bf16* wb = (__bf16*)ws;
  __bf16* fw0_Wih_b = wb + 0;        // 2048x512
  __bf16* fw0_Whh_b = wb + 1048576;  // 2048x512
  __bf16* bw0_Wih_b = wb + 2097152;
  __bf16* bw0_Whh_b = wb + 3145728;
  __bf16* fw1_Wih_b = wb + 4194304;  // 2048x1024
  __bf16* fw1_Whh_b = wb + 6291456;  // 2048x512
  __bf16* bw1_Wih_b = wb + 7340032;  // 2048x1024
  __bf16* bw1_Whh_b = wb + 9437184;  // 2048x512
  __bf16* x0b  = (__bf16*)(ws + 20971520);   // [S,B,512]  bf16 (33.5 MB)
  __bf16* x1b  = (__bf16*)(ws + 54525952);   // [S,B,1024] bf16 (67 MB)
  float*  xgfw = (float*)(ws + 121634816);   // [S*B,2048] f32 (268 MB)
  float*  xgbw = (float*)(ws + 390070272);   // [S*B,2048] f32 (268 MB)
  __bf16* hglob = (__bf16*)(ws + 658505728); // [2][2][64][512] bf16 (256 KB)
  int*    cnt   = (int*)(ws + 658767872);    // [2] barrier counters

  // 1) convert weights to bf16
  cvt_w<<<4096, 256, 0, stream>>>(fw0_Wih, fw0_Wih_b, 1048576);
  cvt_w<<<4096, 256, 0, stream>>>(fw0_Whh, fw0_Whh_b, 1048576);
  cvt_w<<<4096, 256, 0, stream>>>(bw0_Wih, bw0_Wih_b, 1048576);
  cvt_w<<<4096, 256, 0, stream>>>(bw0_Whh, bw0_Whh_b, 1048576);
  cvt_w<<<8192, 256, 0, stream>>>(fw1_Wih, fw1_Wih_b, 2097152);
  cvt_w<<<4096, 256, 0, stream>>>(fw1_Whh, fw1_Whh_b, 1048576);
  cvt_w<<<8192, 256, 0, stream>>>(bw1_Wih, bw1_Wih_b, 2097152);
  cvt_w<<<4096, 256, 0, stream>>>(bw1_Whh, bw1_Whh_b, 1048576);

  // 2) transpose+convert inputs: [B,S,IN] f32 -> [S,B,IN] bf16
  cvt_in<<<65536, 256, 0, stream>>>(inputs, x0b);

  float* outF = (float*)d_out;                 // outputs [B,S,2H]
  float* hnb  = outF + 33554432;               // hn [L,B,2H]
  float* cnb  = hnb + 131072;                  // cn [L,B,2H]

  // 3) layer 0: input projections (whole-device WMMA GEMM), then 32-WG scan
  proj_gemm<<<dim3(256, 32, 2), 256, 0, stream>>>(x0b, 512,
      fw0_Wih_b, bw0_Wih_b, xgfw, xgbw);
  prep_scan<<<256, 256, 0, stream>>>((int*)hglob, cnt);
  lstm_scan_mb<<<32, 1024, 0, stream>>>(xgfw, xgbw, fw0_Whh_b, bw0_Whh_b,
      fw0_bih, fw0_bhh, bw0_bih, bw0_bhh, mask, hglob, cnt,
      x1b, nullptr, hnb, cnb);

  // 4) layer 1
  proj_gemm<<<dim3(256, 32, 2), 256, 0, stream>>>(x1b, 1024,
      fw1_Wih_b, bw1_Wih_b, xgfw, xgbw);
  prep_scan<<<256, 256, 0, stream>>>((int*)hglob, cnt);
  lstm_scan_mb<<<32, 1024, 0, stream>>>(xgfw, xgbw, fw1_Whh_b, bw1_Whh_b,
      fw1_bih, fw1_bhh, bw1_bih, bw1_bhh, mask, hglob, cnt,
      nullptr, outF, hnb + 65536, cnb + 65536);
}
